// MultiHeadCausalSelfAttention_17403207483571
// MI455X (gfx1250) — matrix operand order repaired and checked by reading, not compiled, not run on hardware
//
#include <hip/hip_runtime.h>

typedef __attribute__((ext_vector_type(16))) _Float16 v16h;
typedef __attribute__((ext_vector_type(8)))  _Float16 v8h;
typedef __attribute__((ext_vector_type(8)))  float    v8f;

#define WMMA(a, b, c) __builtin_amdgcn_wmma_f32_16x16x32_f16( \
    false, (a), false, (b), (short)0, (c), false, false)

// ---------------------------------------------------------------------------
// Fragment loaders (wave32, 16x16x32 f16 WMMA)
// A-matrix 16x32 (MxK), row-major source with leading dim `ld` (elements):
//   lane L: row m = L%16; halves 0..15 = K[(L/16)*16 .. +15] contiguous.
// B-matrix 32x16 (KxN), loaded from B^T rows (i.e. [n][k] row-major):
//   lane L: col n = L%16; halves 0..15 = K[(L/16)*16 .. +15] contiguous.
// ---------------------------------------------------------------------------
static __device__ __forceinline__ v16h load_a_frag(const _Float16* __restrict__ p0, int ld) {
  const int lane = threadIdx.x & 31;
  const _Float16* p = p0 + (lane & 15) * ld + ((lane >> 4) << 4);
  v8h lo = *(const v8h*)(p);
  v8h hi = *(const v8h*)(p + 8);
  v16h r;
#pragma unroll
  for (int i = 0; i < 8; ++i) { r[i] = lo[i]; r[i + 8] = hi[i]; }
  return r;
}

static __device__ __forceinline__ v16h load_b_frag(const _Float16* __restrict__ p0, int ld) {
  const int lane = threadIdx.x & 31;
  return *(const v16h*)(p0 + (lane & 15) * ld + ((lane >> 4) << 4));
}

// ---------------------------------------------------------------------------
// fp32 -> f16 conversion
// ---------------------------------------------------------------------------
__global__ void cvt_f32_f16(const float* __restrict__ in, _Float16* __restrict__ out, int n) {
  int i = blockIdx.x * 256 + threadIdx.x;
  if (i < n) out[i] = (_Float16)in[i];
}

// ---------------------------------------------------------------------------
// QKV projection: [8192,512] x [1536,512]^T. Wave tile 64x64, block 128x256.
// sec (q/k/v), head h, batch bb are wave-uniform: a 256-wide block stays inside
// one 512-wide section, n0/m0 are 64-aligned. Epilogue: Q scaled 1/8 ->
// [B,H,N,D], K -> [B,H,N,D], V -> transposed [B,H,D,N] with v8h vector stores.
// ---------------------------------------------------------------------------
__global__ __launch_bounds__(256) void qkv_gemm(
    const _Float16* __restrict__ X, const _Float16* __restrict__ W,
    _Float16* __restrict__ Qb, _Float16* __restrict__ Kb, _Float16* __restrict__ Vt) {
  const int K = 512;
  const int wid = threadIdx.x >> 5, lane = threadIdx.x & 31;
  const int m0 = blockIdx.x * 128 + (wid >> 2) * 64;
  const int n0 = blockIdx.y * 256 + (wid & 3) * 64;
  v8f acc[4][4] = {};
  for (int k = 0; k < K; k += 32) {
    __builtin_prefetch(X + m0 * K + k + 64, 0, 1);
    __builtin_prefetch(W + n0 * K + k + 64, 0, 1);
    v16h a[4], b[4];
#pragma unroll
    for (int i = 0; i < 4; ++i) a[i] = load_a_frag(X + (m0 + 16 * i) * K + k, K);
#pragma unroll
    for (int j = 0; j < 4; ++j) b[j] = load_b_frag(W + (n0 + 16 * j) * K + k, K);
#pragma unroll
    for (int i = 0; i < 4; ++i)
#pragma unroll
      for (int j = 0; j < 4; ++j) acc[i][j] = WMMA(a[i], b[j], acc[i][j]);
  }
  const int hi8 = (lane >> 4) << 3, nc = lane & 15;
  const int sec = n0 >> 9;                 // 0:q 1:k 2:v   (wave-uniform)
  const int h = (n0 >> 6) & 7;             // head          (wave-uniform)
  const int bh = (m0 >> 12) * 8 + h;       // batch*H + h   (wave-uniform)
  const int nn0 = m0 & 4095;               // row within batch (wave-uniform)

  if (sec == 2) {
    // V^T: index (bh*64+d)*4096 + nn; nn is contiguous in r -> one v8h store.
#pragma unroll
    for (int j = 0; j < 4; ++j) {
      const int d = 16 * j + nc;
#pragma unroll
      for (int i = 0; i < 4; ++i) {
        v8h pk;
#pragma unroll
        for (int r = 0; r < 8; ++r) pk[r] = (_Float16)acc[i][j][r];
        *(v8h*)(Vt + (bh * 64 + d) * 4096 + nn0 + 16 * i + hi8) = pk;
      }
    }
  } else {
    _Float16* __restrict__ Dst = (sec == 0) ? Qb : Kb;
    const float scl = (sec == 0) ? 0.125f : 1.0f;
#pragma unroll
    for (int j = 0; j < 4; ++j) {
      const int d = 16 * j + nc;
#pragma unroll
      for (int i = 0; i < 4; ++i) {
#pragma unroll
        for (int r = 0; r < 8; ++r) {
          const int nn = nn0 + 16 * i + r + hi8;
          Dst[(bh * 4096 + nn) * 64 + d] = (_Float16)(acc[i][j][r] * scl);
        }
      }
    }
  }
}

// ---------------------------------------------------------------------------
// Flash attention, causal. One wave per 16 query rows; KV streamed 32/step.
// Q pre-scaled. Online softmax in fp32; P transposed C->A layout via LDS.
// ---------------------------------------------------------------------------
__global__ __launch_bounds__(256) void flash_attn(
    const _Float16* __restrict__ Qb, const _Float16* __restrict__ Kb,
    const _Float16* __restrict__ Vt, _Float16* __restrict__ Oh) {
  __shared__ __align__(16) _Float16 pshm[8][16][32];
  const int N = 4096, D = 64, H = 8;
  const int wid = threadIdx.x >> 5, lane = threadIdx.x & 31;
  const int t = blockIdx.x * 8 + wid;       // 4096 wave-tiles total
  const int b = t >> 11;                    // 2048 tiles per batch (8 heads * 256)
  const int rem = t & 2047;
  const int h = rem >> 8;
  const int i0 = (rem & 255) << 4;
  const _Float16* Qp = Qb + (b * H + h) * N * D;
  const _Float16* Kp = Kb + (b * H + h) * N * D;
  const _Float16* Vp = Vt + (b * H + h) * D * N;

  const v16h qa0 = load_a_frag(Qp + i0 * D, D);
  const v16h qa1 = load_a_frag(Qp + i0 * D + 32, D);
  v8f o[4] = {};
  float mrow[8], lrow[8];
#pragma unroll
  for (int r = 0; r < 8; ++r) { mrow[r] = -1e30f; lrow[r] = 0.0f; }
  const int hi8 = (lane >> 4) << 3, nc = lane & 15;

  for (int j0 = 0; j0 < i0 + 16; j0 += 32) {
    v8f s0 = {}, s1 = {};
    s0 = WMMA(qa0, load_b_frag(Kp + j0 * D, D), s0);
    s0 = WMMA(qa1, load_b_frag(Kp + j0 * D + 32, D), s0);
    s1 = WMMA(qa0, load_b_frag(Kp + (j0 + 16) * D, D), s1);
    s1 = WMMA(qa1, load_b_frag(Kp + (j0 + 16) * D + 32, D), s1);
    if (j0 + 31 > i0) {  // causal frontier
#pragma unroll
      for (int r = 0; r < 8; ++r) {
        const int gi = i0 + r + hi8;
        if (j0 + nc > gi)      s0[r] = -1e30f;
        if (j0 + 16 + nc > gi) s1[r] = -1e30f;
      }
    }
#pragma unroll
    for (int r = 0; r < 8; ++r) {
      float v = fmaxf(s0[r], s1[r]);
#pragma unroll
      for (int off = 8; off >= 1; off >>= 1) v = fmaxf(v, __shfl_xor(v, off, 32));
      const float mnew = fmaxf(mrow[r], v);
      const float fr = __expf(mrow[r] - mnew);
      const float p0 = __expf(s0[r] - mnew);
      const float p1 = __expf(s1[r] - mnew);
      float ps = p0 + p1;
#pragma unroll
      for (int off = 8; off >= 1; off >>= 1) ps += __shfl_xor(ps, off, 32);
      lrow[r] = lrow[r] * fr + ps;
      mrow[r] = mnew;
#pragma unroll
      for (int j = 0; j < 4; ++j) o[j][r] *= fr;
      pshm[wid][r + hi8][nc] = (_Float16)p0;
      pshm[wid][r + hi8][nc + 16] = (_Float16)p1;
    }
    asm volatile("s_wait_dscnt 0" ::: "memory");  // wave-private LDS transpose fence
    const v16h pa = load_a_frag(&pshm[wid][0][0], 32);
#pragma unroll
    for (int j = 0; j < 4; ++j)
      o[j] = WMMA(pa, load_b_frag(Vp + (16 * j) * N + j0, N), o[j]);
    asm volatile("s_wait_dscnt 0" ::: "memory");  // loads retire before next-iter stores
  }
#pragma unroll
  for (int j = 0; j < 4; ++j) {
#pragma unroll
    for (int r = 0; r < 8; ++r) {
      const float v = o[j][r] / lrow[r];
      const int gn = i0 + r + hi8;
      Oh[(b * N + gn) * 512 + h * 64 + 16 * j + nc] = (_Float16)v;
    }
  }
}

// ---------------------------------------------------------------------------
// Output projection: [8192,512] x [512,512]^T + bias, fp32 out.
// ---------------------------------------------------------------------------
__global__ __launch_bounds__(256) void proj_gemm(
    const _Float16* __restrict__ A, const _Float16* __restrict__ W,
    const float* __restrict__ bias, float* __restrict__ Y) {
  const int K = 512, NOUT = 512;
  const int wid = threadIdx.x >> 5, lane = threadIdx.x & 31;
  const int m0 = blockIdx.x * 128 + (wid >> 2) * 64;
  const int n0 = blockIdx.y * 256 + (wid & 3) * 64;
  v8f acc[4][4] = {};
  for (int k = 0; k < K; k += 32) {
    __builtin_prefetch(A + m0 * K + k + 64, 0, 1);
    v16h a[4], b[4];
#pragma unroll
    for (int i = 0; i < 4; ++i) a[i] = load_a_frag(A + (m0 + 16 * i) * K + k, K);
#pragma unroll
    for (int j = 0; j < 4; ++j) b[j] = load_b_frag(W + (n0 + 16 * j) * K + k, K);
#pragma unroll
    for (int i = 0; i < 4; ++i)
#pragma unroll
      for (int j = 0; j < 4; ++j) acc[i][j] = WMMA(a[i], b[j], acc[i][j]);
  }
  const int hi8 = (lane >> 4) << 3, nc = lane & 15;
#pragma unroll
  for (int j = 0; j < 4; ++j) {
    const int f = n0 + 16 * j + nc;
    const float bv = bias[f];
#pragma unroll
    for (int i = 0; i < 4; ++i) {
#pragma unroll
      for (int r = 0; r < 8; ++r) {
        const int gm = m0 + 16 * i + r + hi8;
        Y[gm * NOUT + f] = acc[i][j][r] + bv;
      }
    }
  }
}

// ---------------------------------------------------------------------------
extern "C" void kernel_launch(void* const* d_in, const int* in_sizes, int n_in,
                              void* d_out, int out_size, void* d_ws, size_t ws_size,
                              hipStream_t stream) {
  (void)in_sizes; (void)n_in; (void)out_size; (void)ws_size;
  const float* x      = (const float*)d_in[0];  // [2,4096,512]
  const float* w_qkv  = (const float*)d_in[1];  // [1536,512]
  const float* w_proj = (const float*)d_in[2];  // [512,512]
  const float* b_proj = (const float*)d_in[3];  // [512]
  float* out = (float*)d_out;                   // [2,4096,512]

  const int M = 8192, C = 512, F = 1536;
  _Float16* xh     = (_Float16*)d_ws;
  _Float16* wqkvh  = xh + (size_t)M * C;        // 4,194,304
  _Float16* wprojh = wqkvh + (size_t)F * C;     //   786,432
  _Float16* Qb     = wprojh + (size_t)C * C;    //   262,144
  _Float16* Kb     = Qb + (size_t)M * C;        // B*H*N*D == M*C
  _Float16* Vt     = Kb + (size_t)M * C;
  _Float16* Oh     = Vt + (size_t)M * C;        // total ~44 MB of d_ws

  cvt_f32_f16<<<(M * C + 255) / 256, 256, 0, stream>>>(x, xh, M * C);
  cvt_f32_f16<<<(F * C + 255) / 256, 256, 0, stream>>>(w_qkv, wqkvh, F * C);
  cvt_f32_f16<<<(C * C + 255) / 256, 256, 0, stream>>>(w_proj, wprojh, C * C);

  qkv_gemm<<<dim3(M / 128, F / 256), 256, 0, stream>>>(xh, wqkvh, Qb, Kb, Vt);
  flash_attn<<<dim3((2 * 8 * 256) / 8), 256, 0, stream>>>(Qb, Kb, Vt, Oh);
  proj_gemm<<<dim3(M / 128, C / 256), 256, 0, stream>>>(Oh, wprojh, b_proj, out);
}
